// SelfAttn_MultiHead_30520037606031
// MI455X (gfx1250) — matrix operand order repaired and checked by reading, compiled, not run on hardware
//
#include <hip/hip_runtime.h>

typedef __attribute__((ext_vector_type(16))) __bf16 v16bf;
typedef __attribute__((ext_vector_type(8)))  __bf16 v8bf;
typedef __attribute__((ext_vector_type(8)))  float  v8f;

#define B_   32
#define L_   1024
#define EMB_ 512
#define OUT_ 64
#define H_   8

__device__ __forceinline__ __bf16 f2bf(float f) {
  union { float f; unsigned u; } x; x.f = f;
  unsigned r = x.u + 0x7FFFu + ((x.u >> 16) & 1u);   // round-to-nearest-even
  unsigned short hs = (unsigned short)(r >> 16);
  return __builtin_bit_cast(__bf16, hs);
}

__device__ __forceinline__ v16bf pack16(v8bf lo, v8bf hi) {
  v16bf a;
#pragma unroll
  for (int i = 0; i < 8; ++i) { a[i] = lo[i]; a[i + 8] = hi[i]; }
  return a;
}

// CDNA5 async global->LDS copy (ASYNCcnt-tracked). lds = workgroup LDS byte
// offset (low 32 bits of the flat shared address per the aperture rules).
__device__ __forceinline__ void async_b128(unsigned lds, const char* g) {
  asm volatile("global_load_async_to_lds_b128 %0, %1, off"
               : : "v"(lds), "v"(g) : "memory");
}
__device__ __forceinline__ unsigned lds_off(const void* p) {
  return (unsigned)(unsigned long long)p;   // flat shared addr low 32 bits = LDS offset
}

// ---------------------------------------------------------------- converts
__global__ __launch_bounds__(256) void cvt_joint(const float* __restrict__ x,
                                                 __bf16* __restrict__ y, int n) {
  int i = blockIdx.x * 256 + threadIdx.x;
  if (i < n) y[i] = f2bf(x[i]);
}

// Wq/Wk/Wv [H,512,64] -> [H,64,512] bf16 ; Wo [512,64] -> [64,512] bf16
__global__ __launch_bounds__(256) void cvt_weights(
    const float* __restrict__ Wq, const float* __restrict__ Wk,
    const float* __restrict__ Wv, const float* __restrict__ Wo,
    __bf16* __restrict__ Wqt, __bf16* __restrict__ Wkt,
    __bf16* __restrict__ Wvt, __bf16* __restrict__ Wot) {
  int tid = blockIdx.x * 256 + threadIdx.x;
  const int perW = H_ * OUT_ * EMB_;          // 262144
  if (tid < 3 * perW) {
    int which = tid / perW;
    int r = tid % perW;                        // (h*OUT + o)*EMB + e
    int h = r / (OUT_ * EMB_);
    int o = (r / EMB_) % OUT_;
    int e = r % EMB_;
    const float* src = which == 0 ? Wq : which == 1 ? Wk : Wv;
    __bf16* dst = which == 0 ? Wqt : which == 1 ? Wkt : Wvt;
    dst[r] = f2bf(src[(h * EMB_ + e) * OUT_ + o]);
  } else {
    int t2 = tid - 3 * perW;
    if (t2 < OUT_ * EMB_) {
      int o = t2 / EMB_, e = t2 % EMB_;
      Wot[t2] = f2bf(Wo[e * OUT_ + o]);
    }
  }
}

// ---------------------------------------------------------------- QKV GEMM
// grid (B*16, H, 3), block 128 (4 waves). Wave: 16x64 tile of [1024,512]x[512,64].
__global__ __launch_bounds__(128) void qkv_gemm(
    const __bf16* __restrict__ X,
    const __bf16* __restrict__ Wqt, const __bf16* __restrict__ Wkt,
    const __bf16* __restrict__ Wvt,
    const float* __restrict__ bq, const float* __restrict__ bk,
    const float* __restrict__ bv,
    __bf16* __restrict__ Q, __bf16* __restrict__ K, __bf16* __restrict__ Vt) {
  const int lane = threadIdx.x & 31;
  const int w = threadIdx.x >> 5;
  const int ln = lane & 15, hf = lane >> 4;
  const int b = blockIdx.x >> 4;
  const int m0 = (blockIdx.x & 15) * 64 + w * 16;
  const int h = blockIdx.y;
  const int which = blockIdx.z;
  const __bf16* Wt = which == 0 ? Wqt : which == 1 ? Wkt : Wvt;
  const float* bias = which == 0 ? bq : which == 1 ? bk : bv;

  const __bf16* arow = X + (size_t)(b * L_ + m0 + ln) * EMB_;
  v8f acc[4] = {};
#pragma unroll 4
  for (int kc = 0; kc < 16; ++kc) {
    v16bf a = pack16(*(const v8bf*)(arow + kc * 32 + 16 * hf),
                     *(const v8bf*)(arow + kc * 32 + 16 * hf + 8));
#pragma unroll
    for (int t = 0; t < 4; ++t) {
      const __bf16* wrow =
          Wt + (size_t)(h * OUT_ + t * 16 + ln) * EMB_ + kc * 32 + 16 * hf;
      v16bf bfr = *(const v16bf*)wrow;
      acc[t] = __builtin_amdgcn_wmma_f32_16x16x32_bf16(false, a, false, bfr,
                                                       (short)0, acc[t], false, false);
    }
  }
  const int bh = b * H_ + h;
#pragma unroll
  for (int t = 0; t < 4; ++t) {
    int n = t * 16 + ln;
    float bb = bias[h * OUT_ + n];
#pragma unroll
    for (int r = 0; r < 8; ++r) {
      int row = m0 + r + 8 * hf;
      float val = acc[t][r] + bb;
      if (which == 0)      Q[((size_t)bh * L_ + row) * OUT_ + n] = f2bf(val);
      else if (which == 1) K[((size_t)bh * L_ + row) * OUT_ + n] = f2bf(val);
      else                 Vt[((size_t)bh * OUT_ + n) * L_ + row] = f2bf(val);
    }
  }
}

// ---------------------------------------------------------------- attention
// Block = 4 waves sharing one (b,h); each wave owns one 16-query tile.
// 32-key chunks of K/V are double-buffered in LDS via async global->LDS copies.
__global__ __launch_bounds__(128) void attn(
    const __bf16* __restrict__ Qb, const __bf16* __restrict__ Kb,
    const __bf16* __restrict__ Vtb, const long long* __restrict__ traj_len,
    __bf16* __restrict__ Ob) {
  __shared__ __align__(32) __bf16 KB[2][32 * 64];   // [key][dim], rows 128 B
  __shared__ __align__(32) __bf16 VB[2][64 * 32];   // [dim][key], rows 64 B
  __shared__ __align__(32) __bf16 Pt[4][16 * 32];   // per-wave P tiles

  const int tid = threadIdx.x;
  const int lane = tid & 31;
  const int w = tid >> 5;
  const int ln = lane & 15, hf = lane >> 4;
  const int bh = blockIdx.x >> 4;
  const int q0 = ((blockIdx.x & 15) * 4 + w) * 16;
  const int b = bh >> 3, h = bh & 7;
  const int tl = (int)traj_len[b];
  const float scale = 0.044194173824159216f;   // 1/sqrt(512)

  const char* Kg = (const char*)(Kb + (size_t)bh * L_ * OUT_);    // [key][64] rows 128 B
  const char* Vg = (const char*)(Vtb + (size_t)bh * OUT_ * L_);   // [dim][1024] rows 2048 B
  const int vrow = tid >> 1, vhalf = tid & 1;
  const unsigned kbo[2] = { lds_off(&KB[0][0]) + (unsigned)tid * 32u,
                            lds_off(&KB[1][0]) + (unsigned)tid * 32u };
  const unsigned vbo[2] = { lds_off(&VB[0][0]) + (unsigned)(vrow * 64 + vhalf * 32),
                            lds_off(&VB[1][0]) + (unsigned)(vrow * 64 + vhalf * 32) };

  // Q fragments, loaded once per wave
  const __bf16* qrow = Qb + (size_t)(bh * L_ + q0 + ln) * OUT_;
  v16bf a0 = pack16(*(const v8bf*)(qrow + 16 * hf),      *(const v8bf*)(qrow + 8 + 16 * hf));
  v16bf a1 = pack16(*(const v8bf*)(qrow + 32 + 16 * hf), *(const v8bf*)(qrow + 40 + 16 * hf));

  bool qv[8];
#pragma unroll
  for (int r = 0; r < 8; ++r) qv[r] = (q0 + r + 8 * hf) < tl;

  v8f acc[4] = {};
  float m[8], lsum[8];
#pragma unroll
  for (int r = 0; r < 8; ++r) { m[r] = -1e30f; lsum[r] = 0.f; }

  __bf16* pt = &Pt[w][0];

  // prologue: stage chunk 0 into buffer 0 (4 async b128 per thread)
  {
    const char* kg = Kg + tid * 32;
    const char* vg = Vg + (size_t)vrow * 2048 + vhalf * 32;
    async_b128(kbo[0],      kg);      async_b128(kbo[0] + 16, kg + 16);
    async_b128(vbo[0],      vg);      async_b128(vbo[0] + 16, vg + 16);
  }

  for (int kc = 0; kc < 32; ++kc) {
    const int k0 = kc * 32;
    const int cur = kc & 1, nxt = cur ^ 1;
    if (kc < 31) {     // prefetch next chunk into the other buffer
      const char* kg = Kg + (size_t)(k0 + 32) * 128 + tid * 32;
      const char* vg = Vg + (size_t)vrow * 2048 + (size_t)(k0 + 32) * 2 + vhalf * 32;
      async_b128(kbo[nxt],      kg);      async_b128(kbo[nxt] + 16, kg + 16);
      async_b128(vbo[nxt],      vg);      async_b128(vbo[nxt] + 16, vg + 16);
      asm volatile("s_wait_asynccnt 0x4" ::: "memory");  // cur chunk done (in-order)
    } else {
      asm volatile("s_wait_asynccnt 0x0" ::: "memory");
    }
    __syncthreads();   // all waves' copies for cur chunk visible

    const __bf16* kbuf = &KB[cur][0];
    const __bf16* vbuf = &VB[cur][0];

    v8f S[2];
#pragma unroll
    for (int j = 0; j < 2; ++j) {
      const __bf16* krow = kbuf + (j * 16 + ln) * 64;
      v16bf b0 = *(const v16bf*)(krow + 16 * hf);
      v16bf b1 = *(const v16bf*)(krow + 32 + 16 * hf);
      v8f z = {};
      z = __builtin_amdgcn_wmma_f32_16x16x32_bf16(false, a0, false, b0, (short)0, z, false, false);
      S[j] = __builtin_amdgcn_wmma_f32_16x16x32_bf16(false, a1, false, b1, (short)0, z, false, false);
    }
    const bool kv0 = (k0 + ln) < tl;
    const bool kv1 = (k0 + 16 + ln) < tl;

    float p0[8], p1[8], mt[8];
#pragma unroll
    for (int r = 0; r < 8; ++r) {
      float s0 = (kv0 && qv[r]) ? S[0][r] * scale : 0.f;   // multiplicative mask
      float s1 = (kv1 && qv[r]) ? S[1][r] * scale : 0.f;
      p0[r] = s0; p1[r] = s1;
      mt[r] = fmaxf(s0, s1);
    }
#pragma unroll
    for (int r = 0; r < 8; ++r)
#pragma unroll
      for (int off = 1; off < 16; off <<= 1)
        mt[r] = fmaxf(mt[r], __shfl_xor(mt[r], off, 32));

#pragma unroll
    for (int r = 0; r < 8; ++r) {
      float mn = fmaxf(m[r], mt[r]);
      float corr = __expf(m[r] - mn);
      m[r] = mn;
      p0[r] = __expf(p0[r] - mn);
      p1[r] = __expf(p1[r] - mn);
      lsum[r] *= corr;
#pragma unroll
      for (int t = 0; t < 4; ++t) acc[t][r] *= corr;
    }
#pragma unroll
    for (int r = 0; r < 8; ++r) {
      float s = p0[r] + p1[r];
#pragma unroll
      for (int off = 1; off < 16; off <<= 1) s += __shfl_xor(s, off, 32);
      lsum[r] += s;
    }
    // re-layout P (C-frag) -> A-frag via wave-private LDS tile
#pragma unroll
    for (int r = 0; r < 8; ++r) {
      pt[(r + 8 * hf) * 32 + ln]      = f2bf(p0[r]);
      pt[(r + 8 * hf) * 32 + 16 + ln] = f2bf(p1[r]);
    }
    __builtin_amdgcn_wave_barrier();
    v16bf ap = pack16(*(const v8bf*)(pt + ln * 32 + 16 * hf),
                      *(const v8bf*)(pt + ln * 32 + 8 + 16 * hf));
    __builtin_amdgcn_wave_barrier();
#pragma unroll
    for (int t = 0; t < 4; ++t) {
      v16bf bv_ = *(const v16bf*)(vbuf + (t * 16 + ln) * 32 + 16 * hf);
      acc[t] = __builtin_amdgcn_wmma_f32_16x16x32_bf16(false, ap, false, bv_,
                                                       (short)0, acc[t], false, false);
    }
    __syncthreads();   // all reads of cur buffer done before it is overwritten
  }
#pragma unroll
  for (int t = 0; t < 4; ++t)
#pragma unroll
    for (int r = 0; r < 8; ++r) {
      float val = acc[t][r] / lsum[r];
      Ob[(size_t)(b * L_ + q0 + r + 8 * hf) * (H_ * OUT_) + h * OUT_ + t * 16 + ln] =
          f2bf(val);
    }
}

// ---------------------------------------------------------------- out proj
// [32768,512] x [512,64] -> d_out f32
__global__ __launch_bounds__(128) void out_gemm(const __bf16* __restrict__ Ob,
                                                const __bf16* __restrict__ Wot,
                                                float* __restrict__ out) {
  const int lane = threadIdx.x & 31;
  const int w = threadIdx.x >> 5;
  const int ln = lane & 15, hf = lane >> 4;
  const int m0 = blockIdx.x * 64 + w * 16;
  const __bf16* arow = Ob + (size_t)(m0 + ln) * 512;
  v8f acc[4] = {};
#pragma unroll 4
  for (int kc = 0; kc < 16; ++kc) {
    v16bf a = pack16(*(const v8bf*)(arow + kc * 32 + 16 * hf),
                     *(const v8bf*)(arow + kc * 32 + 16 * hf + 8));
#pragma unroll
    for (int t = 0; t < 4; ++t) {
      v16bf bfr = *(const v16bf*)(Wot + (size_t)(t * 16 + ln) * 512 + kc * 32 + 16 * hf);
      acc[t] = __builtin_amdgcn_wmma_f32_16x16x32_bf16(false, a, false, bfr,
                                                       (short)0, acc[t], false, false);
    }
  }
#pragma unroll
  for (int t = 0; t < 4; ++t)
#pragma unroll
    for (int r = 0; r < 8; ++r)
      out[(size_t)(m0 + r + 8 * hf) * 64 + t * 16 + ln] = acc[t][r];
}

// ---------------------------------------------------------------- launch
extern "C" void kernel_launch(void* const* d_in, const int* in_sizes, int n_in,
                              void* d_out, int out_size, void* d_ws, size_t ws_size,
                              hipStream_t stream) {
  (void)in_sizes; (void)n_in; (void)out_size; (void)ws_size;
  const float*     joint = (const float*)d_in[0];
  // d_in[1] = delta: only defines mask shape in the reference; never read.
  const long long* traj  = (const long long*)d_in[2];
  const float* Wq = (const float*)d_in[3];
  const float* Wk = (const float*)d_in[4];
  const float* Wv = (const float*)d_in[5];
  const float* bq = (const float*)d_in[6];
  const float* bk = (const float*)d_in[7];
  const float* bv = (const float*)d_in[8];
  const float* Wo = (const float*)d_in[9];

  char* ws = (char*)d_ws;
  __bf16* Qb  = (__bf16*)(ws);                 // 32 MB  [B,H,L,64]
  __bf16* Kb  = (__bf16*)(ws + 33554432);      // 32 MB  [B,H,L,64]
  __bf16* Vtb = (__bf16*)(ws + 67108864);      // 32 MB  [B,H,64,L]
  __bf16* Jb  = (__bf16*)(ws + 100663296);     // 32 MB  [B,L,512]; reused as Ob
  __bf16* Ob  = Jb;
  __bf16* Wqt = (__bf16*)(ws + 134217728);     // 512 KB [H,64,512]
  __bf16* Wkt = (__bf16*)(ws + 134742016);
  __bf16* Wvt = (__bf16*)(ws + 135266304);
  __bf16* Wot = (__bf16*)(ws + 135790592);     // 64 KB  [64,512]

  cvt_joint<<<65536, 256, 0, stream>>>(joint, Jb, B_ * L_ * EMB_);
  cvt_weights<<<3200, 256, 0, stream>>>(Wq, Wk, Wv, Wo, Wqt, Wkt, Wvt, Wot);
  qkv_gemm<<<dim3(512, 8, 3), 128, 0, stream>>>(Jb, Wqt, Wkt, Wvt, bq, bk, bv,
                                                Qb, Kb, Vtb);
  attn<<<4096, 128, 0, stream>>>(Qb, Kb, Vtb, traj, Ob);
  out_gemm<<<512, 128, 0, stream>>>(Ob, Wot, (float*)d_out);
}
